// LowGraph_3539053052649
// MI455X (gfx1250) — matrix operand order repaired and checked by reading, not compiled, not run on hardware
//
#include <hip/hip_runtime.h>
#include <hip/hip_bf16.h>
#include <math.h>

typedef __attribute__((ext_vector_type(16))) __bf16 v16bf;
typedef __attribute__((ext_vector_type(8)))  __bf16 v8bf;
typedef __attribute__((ext_vector_type(4)))  __bf16 v4bf;
typedef __attribute__((ext_vector_type(8)))  float  v8f;

#define BS    256
#define NN    200
#define DIN   320
#define F1    32
#define H1    64      // concat of 2 heads
#define F2    64
#define NROWS (BS*NN) // 51200
#define NPADM 208     // 13 * 16
#define NPADK 224     // 7  * 32
#define GAT_ALPHA 0.2f

// ---------------- WMMA fragment loaders (wave32, contiguous k order, same in A and B) -------
// 16-bit A-matrix 16x32 (MxK): lane -> m = lane&15, half = lane>>4.
// Per lane the 16 elements are one contiguous run:
//   e in [0,16) : k = half*16 + e
__device__ inline v16bf load_A_frag(const __bf16* base, int row0, int k0,
                                    int ld, int lane) {
  int m = lane & 15, half = lane >> 4;
  const __bf16* p = base + (size_t)(row0 + m) * ld + k0 + half * 16;
  v8bf a0 = *(const v8bf*)p;          // 16B
  v8bf a1 = *(const v8bf*)(p + 8);    // 16B
  return __builtin_shufflevector(a0, a1, 0, 1, 2, 3, 4, 5, 6, 7,
                                 8, 9, 10, 11, 12, 13, 14, 15);
}

// 16-bit B-matrix 32x16 (KxN) from N-MAJOR (transposed) storage baseT[n][k]:
// lane -> n = lane&15; element e -> k = half*16 + e  (one contiguous run).
__device__ inline v16bf load_B_fragT(const __bf16* baseT, int k0, int n0,
                                     int ld, int lane) {
  int n = lane & 15, half = lane >> 4;
  const __bf16* p = baseT + (size_t)(n0 + n) * ld + k0 + half * 16;
  v8bf b0 = *(const v8bf*)p;
  v8bf b1 = *(const v8bf*)(p + 8);
  return __builtin_shufflevector(b0, b1, 0, 1, 2, 3, 4, 5, 6, 7,
                                 8, 9, 10, 11, 12, 13, 14, 15);
}

// ---------------- prep: weights -> bf16 (N-major), adj -> bitmask -----------

__global__ void prep_kernel(const float* __restrict__ W1_0,
                            const float* __restrict__ W1_1,
                            const float* __restrict__ W2,
                            const int* __restrict__ adj,
                            __bf16* __restrict__ Wb1T,   // [H1][DIN]
                            __bf16* __restrict__ Wb2T,   // [F2][H1]
                            unsigned* __restrict__ adjbits) {
  int tid = threadIdx.x;
  for (int t = tid; t < DIN * H1; t += blockDim.x) {
    int n = t / DIN, k = t % DIN;
    float w = (n < F1) ? W1_0[k * F1 + n] : W1_1[k * F1 + (n - F1)];
    Wb1T[t] = (__bf16)w;
  }
  for (int t = tid; t < H1 * F2; t += blockDim.x) {
    int n = t / H1, k = t % H1;
    Wb2T[t] = (__bf16)W2[k * F2 + n];
  }
  for (int t = tid; t < NN * 7; t += blockDim.x) {
    int i = t / 7, wj = t % 7;
    unsigned bits = 0u;
    for (int bb = 0; bb < 32; ++bb) {
      int j = wj * 32 + bb;
      if (j < NN && adj[i * NN + j] > 0) bits |= (1u << bb);
    }
    adjbits[t] = bits;
  }
}

// ---------------- GEMM: h = X(bf16) @ Wb(bf16) -> f32, N = 64 ---------------
// block = 128 threads (4 waves); 16 rows per block; wave w owns N-tile w.
// Staging: one float4 load + one ds_store_b64 per thread per k-chunk.
__global__ __launch_bounds__(128)
void gemm_kernel(const float* __restrict__ xin, int ldx, int kchunks,
                 const __bf16* __restrict__ WbT, int ldw,
                 float* __restrict__ hout) {
  __shared__ __align__(16) __bf16 xa[16][32];
  int tid = threadIdx.x, lane = tid & 31, wave = tid >> 5;
  int row0 = blockIdx.x * 16;
  int r = (tid * 4) >> 5, kk = (tid * 4) & 31;   // this thread's 4 elements
  v8f acc = {};
  for (int kc = 0; kc < kchunks; ++kc) {
    float4 vx = *(const float4*)(xin + (size_t)(row0 + r) * ldx + kc * 32 + kk);
    v4bf w;
    w[0] = (__bf16)vx.x; w[1] = (__bf16)vx.y;
    w[2] = (__bf16)vx.z; w[3] = (__bf16)vx.w;
    *(v4bf*)&xa[r][kk] = w;
    __syncthreads();
    v16bf afrag = load_A_frag(&xa[0][0], 0, 0, 32, lane);
    v16bf bfrag = load_B_fragT(WbT, kc * 32, wave * 16, ldw, lane);
    acc = __builtin_amdgcn_wmma_f32_16x16x32_bf16(
        false, afrag, false, bfrag, (short)0, acc, false, false);
    __syncthreads();
  }
  int n = lane & 15;
#pragma unroll
  for (int rr = 0; rr < 8; ++rr) {
    int m = rr + ((lane >> 4) << 3);
    hout[(size_t)(row0 + m) * H1 + wave * 16 + n] = acc[rr];
  }
}

// ---------------- per-node attention logits src/dst -------------------------

__global__ void srcdst1_kernel(const float* __restrict__ h1,
                               const float* __restrict__ a1_0,
                               const float* __restrict__ a1_1,
                               float* __restrict__ sd) {
  __shared__ float a0s[2 * F1], a1s[2 * F1];
  int tid = threadIdx.x;
  if (tid < 2 * F1) { a0s[tid] = a1_0[tid]; a1s[tid] = a1_1[tid]; }
  __syncthreads();
  int row = blockIdx.x * blockDim.x + tid;
  if (row >= NROWS) return;
  const float* hr = h1 + (size_t)row * H1;
  float s0 = 0.f, d0 = 0.f, s1 = 0.f, d1 = 0.f;
#pragma unroll 4
  for (int f = 0; f < F1; ++f) {
    float v0 = hr[f], v1 = hr[F1 + f];
    s0 += v0 * a0s[f];      d0 += v0 * a0s[F1 + f];
    s1 += v1 * a1s[f];      d1 += v1 * a1s[F1 + f];
  }
  sd[row] = s0; sd[NROWS + row] = d0;
  sd[2 * (size_t)NROWS + row] = s1; sd[3 * (size_t)NROWS + row] = d1;
}

__global__ void srcdst2_kernel(const float* __restrict__ h2,
                               const float* __restrict__ a2,
                               float* __restrict__ sd2) {
  __shared__ float as[2 * F2];
  int tid = threadIdx.x;
  if (tid < 2 * F2) as[tid] = a2[tid];
  __syncthreads();
  int row = blockIdx.x * blockDim.x + tid;
  if (row >= NROWS) return;
  const float* hr = h2 + (size_t)row * F2;
  float s = 0.f, d = 0.f;
#pragma unroll 4
  for (int f = 0; f < F2; ++f) {
    float v = hr[f];
    s += v * as[f]; d += v * as[F2 + f];
  }
  sd2[row] = s; sd2[NROWS + row] = d;
}

// ---------------- attention: masked softmax + att@h + ELU -------------------
// One block = one (graph, head). 256 threads = 8 waves. att (bf16) and the
// transposed h operand live in LDS; softmax normalization folded into the
// WMMA epilogue via f32 reciprocal row sums.
template <int F>
__global__ __launch_bounds__(256)
void attn_kernel(const float* __restrict__ h, const float* __restrict__ sd,
                 const unsigned* __restrict__ adjbits, float* __restrict__ out,
                 int outbase) {
  extern __shared__ char smem[];
  __bf16* att    = (__bf16*)smem;                          // [208][224]
  __bf16* hBt    = (__bf16*)(smem + NPADM * NPADK * 2);    // [F][224] N-major
  float* rowinv  = (float*)(smem + NPADM * NPADK * 2 + (size_t)F * NPADK * 2);
  float* srcl    = rowinv + NPADM;
  float* dstl    = srcl + NPADM;
  unsigned* adjl = (unsigned*)(dstl + NPADK);              // [200*7]

  int tid = threadIdx.x, lane = tid & 31, wave = tid >> 5;
  int b = blockIdx.x, head = blockIdx.y;
  const float* srcg = sd + (size_t)(2 * head) * NROWS + (size_t)b * NN;
  const float* dstg = sd + (size_t)(2 * head + 1) * NROWS + (size_t)b * NN;

  // stage h (transposed to N-major) -- coalesced reads along f
  for (int t = tid; t < NPADK * F; t += blockDim.x) {
    int j = t / F, f = t % F;
    hBt[(size_t)f * NPADK + j] =
        (j < NN) ? (__bf16)h[((size_t)b * NN + j) * H1 + head * F + f]
                 : (__bf16)0.0f;
  }
  for (int t = tid; t < NPADK; t += blockDim.x)
    dstl[t] = (t < NN) ? dstg[t] : 0.0f;
  for (int t = tid; t < NPADM; t += blockDim.x)
    srcl[t] = (t < NN) ? srcg[t] : 0.0f;
  for (int t = tid; t < NN * 7; t += blockDim.x) adjl[t] = adjbits[t];
  __syncthreads();

  // rows: pass1 max, pass2 exp(e-max) stored bf16; keep 1/rowsum in f32
  for (int i = tid; i < NPADM; i += blockDim.x) {
    if (i < NN) {
      float si = srcl[i];
      float mx = -3.0e38f;
      for (int j = 0; j < NN; ++j) {
        if (adjl[i * 7 + (j >> 5)] & (1u << (j & 31))) {
          float v = si + dstl[j];
          float lr = v > 0.f ? v : GAT_ALPHA * v;
          mx = fmaxf(mx, lr);
        }
      }
      if (mx < -1.0e38f) {          // isolated node: softmax of NEG_INF row
        for (int j = 0; j < NPADK; ++j)
          att[i * NPADK + j] = (__bf16)((j < NN) ? 1.0f : 0.0f);
        rowinv[i] = 1.0f / (float)NN;
      } else {
        float s = 0.f;
        for (int j = 0; j < NPADK; ++j) {
          float p = 0.f;
          if (j < NN && (adjl[i * 7 + (j >> 5)] & (1u << (j & 31)))) {
            float v = si + dstl[j];
            float lr = v > 0.f ? v : GAT_ALPHA * v;
            p = __expf(lr - mx);
          }
          s += p;
          att[i * NPADK + j] = (__bf16)p;
        }
        rowinv[i] = 1.0f / s;
      }
    } else {
      for (int j = 0; j < NPADK; ++j) att[i * NPADK + j] = (__bf16)0.0f;
      rowinv[i] = 1.0f;
    }
  }
  __syncthreads();

  // att(208x224) @ h(224xF): 13 M-tiles x (F/16) N-tiles over 8 waves
  const int MT = NPADM / 16, NT = F / 16;
  for (int job = wave; job < MT * NT; job += 8) {
    int mt = job % MT, nt = job / MT;
    v8f acc = {};
#pragma unroll
    for (int kc = 0; kc < NPADK / 32; ++kc) {
      v16bf afrag = load_A_frag(att, mt * 16, kc * 32, NPADK, lane);
      v16bf bfrag = load_B_fragT(hBt, kc * 32, nt * 16, NPADK, lane);
      acc = __builtin_amdgcn_wmma_f32_16x16x32_bf16(
          false, afrag, false, bfrag, (short)0, acc, false, false);
    }
    int n = lane & 15;
#pragma unroll
    for (int rr = 0; rr < 8; ++rr) {
      int m = rr + ((lane >> 4) << 3);
      int node = mt * 16 + m;
      if (node < NN) {
        float v = acc[rr] * rowinv[node];
        float ev = v > 0.f ? v : (__expf(v) - 1.0f);    // ELU (alpha=1)
        out[((size_t)b * NN + node) * 128 + outbase + head * F + nt * 16 + n]
            = ev;
      }
    }
  }
}

// ---------------- host launcher ---------------------------------------------

extern "C" void kernel_launch(void* const* d_in, const int* in_sizes, int n_in,
                              void* d_out, int out_size, void* d_ws,
                              size_t ws_size, hipStream_t stream) {
  const float* x    = (const float*)d_in[0];
  const int*   adj  = (const int*)d_in[1];
  const float* W1_0 = (const float*)d_in[2];
  const float* a1_0 = (const float*)d_in[3];
  const float* W1_1 = (const float*)d_in[4];
  const float* a1_1 = (const float*)d_in[5];
  const float* W2_0 = (const float*)d_in[6];
  const float* a2_0 = (const float*)d_in[7];
  float* out = (float*)d_out;

  char* ws = (char*)d_ws;
  size_t off = 0;
  auto carve = [&](size_t bytes) -> void* {
    void* p = (void*)(ws + off);
    off = (off + bytes + 255) & ~(size_t)255;
    return p;
  };
  __bf16*   Wb1T    = (__bf16*)carve((size_t)DIN * H1 * sizeof(__bf16));
  __bf16*   Wb2T    = (__bf16*)carve((size_t)H1 * F2 * sizeof(__bf16));
  unsigned* adjbits = (unsigned*)carve((size_t)NN * 7 * sizeof(unsigned));
  float*    sd1     = (float*)carve((size_t)4 * NROWS * sizeof(float));
  float*    sd2     = (float*)carve((size_t)2 * NROWS * sizeof(float));
  float*    h1      = (float*)carve((size_t)NROWS * H1 * sizeof(float));
  float*    h2      = (float*)carve((size_t)NROWS * F2 * sizeof(float));

  prep_kernel<<<1, 256, 0, stream>>>(W1_0, W1_1, W2_0, adj, Wb1T, Wb2T,
                                     adjbits);

  // layer 1: h1 = x @ [W1_0 | W1_1]
  gemm_kernel<<<NROWS / 16, 128, 0, stream>>>(x, DIN, DIN / 32, Wb1T, DIN, h1);
  srcdst1_kernel<<<NROWS / 256, 256, 0, stream>>>(h1, a1_0, a1_1, sd1);

  size_t sh1 = (size_t)NPADM * NPADK * 2 + (size_t)F1 * NPADK * 2 +
               (2 * NPADM + NPADK) * sizeof(float) + NN * 7 * sizeof(unsigned);
  attn_kernel<F1><<<dim3(BS, 2), 256, sh1, stream>>>(h1, sd1, adjbits, out, 0);

  // layer 2: h2 = x1 @ W2 (x1 read back from out[...,0:64])
  gemm_kernel<<<NROWS / 16, 128, 0, stream>>>((const float*)d_out, 128,
                                              H1 / 32, Wb2T, H1, h2);
  srcdst2_kernel<<<NROWS / 256, 256, 0, stream>>>(h2, a2_0, sd2);

  size_t sh2 = (size_t)NPADM * NPADK * 2 + (size_t)F2 * NPADK * 2 +
               (2 * NPADM + NPADK) * sizeof(float) + NN * 7 * sizeof(unsigned);
  attn_kernel<F2><<<dim3(BS, 1), 256, sh2, stream>>>(h2, sd2, adjbits, out, H1);
}
